// SelfMultiHeadAttention_58368605552963
// MI455X (gfx1250) — matrix operand order repaired and checked by reading, not compiled, not run on hardware
//
#include <hip/hip_runtime.h>

typedef __attribute__((ext_vector_type(16))) _Float16 v16h;
typedef __attribute__((ext_vector_type(8)))  _Float16 v8h;
typedef __attribute__((ext_vector_type(4)))  _Float16 v4h;
typedef __attribute__((ext_vector_type(8)))  float    v8f;

#define BLn   16      // b*l
#define NSEQ  512     // n
#define EDIM  1024    // e
#define NH    16      // heads
#define DH    64      // head dim
#define N3E   3072    // 3*e

// Build a v16h A/B fragment from two 16-byte LDS/global chunks.
__device__ __forceinline__ v16h ld16pair(const _Float16* p0, const _Float16* p1) {
  v8h a = *(const v8h*)p0;
  v8h b = *(const v8h*)p1;
  v16h r;
#pragma unroll
  for (int i = 0; i < 8; ++i) { r[i] = a[i]; r[i + 8] = b[i]; }
  return r;
}
__device__ __forceinline__ v16h ld16(const _Float16* p) { return ld16pair(p, p + 8); }

__device__ __forceinline__ v8f wmma16(v16h a, v16h b, v8f c) {
  return __builtin_amdgcn_wmma_f32_16x16x32_f16(false, a, false, b, (short)0, c,
                                                false, false);
}

// ---------------------------------------------------------------------------
// Kernel 1: qkv = x @ w_qkv + b_qkv  (8192x3072, K=1024), f16 WMMA, f32 accum.
// 64-deep k-slabs staged in LDS; B-fragments batched ahead of the WMMA burst.
// Writes Q*(1/sqrt(d)) and K as (head, n, d) f16; V transposed as (head, d, n).
// ---------------------------------------------------------------------------
__global__ __launch_bounds__(256)
void qkv_gemm_kernel(const float* __restrict__ x, const float* __restrict__ w,
                     const float* __restrict__ bias,
                     _Float16* __restrict__ qf, _Float16* __restrict__ kf,
                     _Float16* __restrict__ vf) {
  // rows padded to 72 halves (144B): 16B-aligned fragment chunks, bank spread
  __shared__ __align__(16) _Float16 sA[128 * 72];   // A tile: [m][k]
  __shared__ __align__(16) _Float16 sB[128 * 72];   // B tile transposed: [c][k]

  const unsigned tid  = threadIdx.x;
  const unsigned wv   = tid >> 5;
  const unsigned lane = tid & 31;
  const unsigned hi   = lane >> 4;        // which 16-lane half
  const unsigned ln   = lane & 15;
  const unsigned tileM = blockIdx.y * 128;
  const unsigned tileN = blockIdx.x * 128;

  v8f acc[8] = {};

  for (unsigned kb = 0; kb < EDIM; kb += 64) {
    __syncthreads();
    // stage A: 128 rows x 64 k, float4 global chunks -> packed 8B LDS stores
#pragma unroll
    for (unsigned p = 0; p < 8; ++p) {
      const unsigned cc = tid + p * 256;
      const unsigned m = cc >> 4, k4 = (cc & 15) << 2;
      const float4 f = *(const float4*)&x[(size_t)(tileM + m) * EDIM + kb + k4];
      v4h h4;
      h4[0] = (_Float16)f.x; h4[1] = (_Float16)f.y;
      h4[2] = (_Float16)f.z; h4[3] = (_Float16)f.w;
      *(v4h*)&sA[m * 72 + k4] = h4;
    }
    // stage B transposed: 64 k x 128 cols, float4 chunks -> 4 b16 LDS stores
#pragma unroll
    for (unsigned p = 0; p < 8; ++p) {
      const unsigned cc = tid + p * 256;
      const unsigned k = cc >> 5, c4 = (cc & 31) << 2;
      const float4 f = *(const float4*)&w[(size_t)(kb + k) * N3E + tileN + c4];
      sB[(c4 + 0) * 72 + k] = (_Float16)f.x;
      sB[(c4 + 1) * 72 + k] = (_Float16)f.y;
      sB[(c4 + 2) * 72 + k] = (_Float16)f.z;
      sB[(c4 + 3) * 72 + k] = (_Float16)f.w;
    }
    __syncthreads();

#pragma unroll
    for (unsigned kh = 0; kh < 2; ++kh) {
      // A fragment: lane holds row (wv*16+ln); halves K = 0..15 + 16*hi
      const _Float16* arow = &sA[(wv * 16 + ln) * 72 + kh * 32];
      const v16h afrag = ld16(arow + 16 * hi);
      // Batch all 8 B fragments, then fire 8 independent WMMAs back-to-back
      v16h bfr[8];
#pragma unroll
      for (unsigned t = 0; t < 8; ++t)
        bfr[t] = ld16(&sB[(t * 16 + ln) * 72 + kh * 32 + 16 * hi]);
#pragma unroll
      for (unsigned t = 0; t < 8; ++t)
        acc[t] = wmma16(afrag, bfr[t], acc[t]);
    }
  }

  // Epilogue: bias, convert, scatter (all divisors are powers of two -> shifts)
#pragma unroll
  for (unsigned t = 0; t < 8; ++t) {
    const unsigned c    = tileN + t * 16 + ln;
    const float bv      = bias[c];
    const unsigned which = c >> 10;        // 0=Q 1=K 2=V
    const unsigned rem   = c & 1023;
    const unsigned hh    = rem >> 6;
    const unsigned dv    = rem & 63;
#pragma unroll
    for (unsigned r = 0; r < 8; ++r) {
      const unsigned rowg = tileM + wv * 16 + r + 8 * hi;
      const unsigned bl   = rowg >> 9;
      const unsigned ni   = rowg & 511;
      const size_t hb = ((size_t)((bl << 4) + hh)) * (NSEQ * DH);
      const float val = acc[t][r] + bv;
      if (which == 0)      qf[hb + (size_t)ni * DH + dv]   = (_Float16)(val * 0.125f);
      else if (which == 1) kf[hb + (size_t)ni * DH + dv]   = (_Float16)val;
      else                 vf[hb + (size_t)dv * NSEQ + ni] = (_Float16)val;
    }
  }
}

// ---------------------------------------------------------------------------
// Kernel 2: flash attention. grid = (n/128 row blocks, b*l*H heads).
// 8 waves x 16 query rows; K/V staged per 64-column block in LDS.
// ---------------------------------------------------------------------------
__global__ __launch_bounds__(256)
void flash_attn_kernel(const _Float16* __restrict__ qf, const _Float16* __restrict__ kf,
                       const _Float16* __restrict__ vf, const float* __restrict__ mask,
                       float* __restrict__ out) {
  __shared__ __align__(16) _Float16 sK[64 * 72];      // [key][d]   (+pad)
  __shared__ __align__(16) _Float16 sV[64 * 72];      // [d][key]   (+pad)
  __shared__ __align__(16) _Float16 sP[8 * 16 * 72];  // per-wave P transpose scratch

  const unsigned tid  = threadIdx.x;
  const unsigned wv   = tid >> 5;
  const unsigned lane = tid & 31;
  const unsigned hi   = lane >> 4;
  const unsigned ln   = lane & 15;

  const unsigned headi = blockIdx.y;              // bl*16 + h
  const unsigned bl    = headi >> 4;
  const unsigned hh    = headi & 15;
  const unsigned rowb  = blockIdx.x * 128 + wv * 16;

  const _Float16* kbase = kf + (size_t)headi * (NSEQ * DH);
  const _Float16* vbase = vf + (size_t)headi * (DH * NSEQ);
  const _Float16* qrow  = qf + (size_t)headi * (NSEQ * DH) + (size_t)(rowb + ln) * DH;

  // Resident Q A-fragments (16 rows x 64 d = two 16x32 A matrices)
  v16h qa[2];
#pragma unroll
  for (unsigned kh = 0; kh < 2; ++kh)
    qa[kh] = ld16(qrow + 32 * kh + 16 * hi);

  v8f o[4] = {};
  v8f rmax, rsum;
#pragma unroll
  for (int r = 0; r < 8; ++r) { rmax[r] = -1e30f; rsum[r] = 0.f; }

  _Float16* pw = &sP[wv * 16 * 72];

  for (unsigned j = 0; j < 8; ++j) {            // 8 blocks of 64 keys
    __syncthreads();
    for (unsigned c = tid; c < 512; c += 256) { // 8KB K + 8KB V, 16B chunks
      const unsigned row = c >> 3, off = c & 7;
      *(v8h*)&sK[row * 72 + off * 8] =
          *(const v8h*)(kbase + (size_t)j * 64 * DH + c * 8);
      *(v8h*)&sV[row * 72 + off * 8] =
          *(const v8h*)(vbase + (size_t)row * NSEQ + j * 64 + off * 8);
    }
    if (j + 1 < 8 && tid < 64)                  // prefetch next K block
      __builtin_prefetch(kbase + (size_t)(j + 1) * 64 * DH + tid * 64, 0, 1);
    __syncthreads();

    // S = (Q/sqrt(d)) @ K^T : batch 8 B fragments, then 8 WMMAs (4 indep chains)
    v16h kfr[8];
#pragma unroll
    for (unsigned t = 0; t < 4; ++t)
#pragma unroll
      for (unsigned kh = 0; kh < 2; ++kh)
        kfr[t * 2 + kh] = ld16(&sK[(t * 16 + ln) * 72 + 32 * kh + 16 * hi]);

    v8f s[4];
#pragma unroll
    for (unsigned t = 0; t < 4; ++t) {
      v8f cacc = {};
      cacc = wmma16(qa[0], kfr[t * 2 + 0], cacc);
      cacc = wmma16(qa[1], kfr[t * 2 + 1], cacc);
      s[t] = cacc;
    }

    // + attention_mask (broadcast over heads)
    const float* mrow = mask + (((size_t)bl * NSEQ + rowb) << 9) + j * 64;
#pragma unroll
    for (unsigned t = 0; t < 4; ++t)
#pragma unroll
      for (unsigned r = 0; r < 8; ++r)
        s[t][r] += mrow[((size_t)(r + 8 * hi) << 9) + t * 16 + ln];

    // online softmax: row max/sum reductions across 16-lane halves
    v8f tmax = s[0];
#pragma unroll
    for (int t = 1; t < 4; ++t)
#pragma unroll
      for (int r = 0; r < 8; ++r) tmax[r] = fmaxf(tmax[r], s[t][r]);
#pragma unroll
    for (int d = 1; d < 16; d <<= 1)
#pragma unroll
      for (int r = 0; r < 8; ++r) tmax[r] = fmaxf(tmax[r], __shfl_xor(tmax[r], d, 32));

    v8f alpha;
#pragma unroll
    for (int r = 0; r < 8; ++r) {
      const float mn = fmaxf(rmax[r], tmax[r]);
      alpha[r] = __expf(rmax[r] - mn);
      rmax[r] = mn;
    }

    v8f psum = {};
#pragma unroll
    for (int t = 0; t < 4; ++t)
#pragma unroll
      for (int r = 0; r < 8; ++r) {
        const float p = __expf(s[t][r] - rmax[r]);
        s[t][r] = p;
        psum[r] += p;
      }
#pragma unroll
    for (int d = 1; d < 16; d <<= 1)
#pragma unroll
      for (int r = 0; r < 8; ++r) psum[r] += __shfl_xor(psum[r], d, 32);
#pragma unroll
    for (int r = 0; r < 8; ++r) rsum[r] = rsum[r] * alpha[r] + psum[r];
#pragma unroll
    for (int t = 0; t < 4; ++t)
#pragma unroll
      for (int r = 0; r < 8; ++r) o[t][r] *= alpha[r];

    // Transpose P (C-layout -> A-layout) through per-wave LDS (DS in-order per wave)
#pragma unroll
    for (unsigned t = 0; t < 4; ++t)
#pragma unroll
      for (unsigned r = 0; r < 8; ++r)
        pw[(r + 8 * hi) * 72 + t * 16 + ln] = (_Float16)s[t][r];
    asm volatile("" ::: "memory");

    // O += P @ V : batch P and V fragments, then 8 WMMAs
    const _Float16* pr = pw + ln * 72;
    v16h pa[2], vfr[8];
#pragma unroll
    for (unsigned kh = 0; kh < 2; ++kh) {
      pa[kh] = ld16(pr + 32 * kh + 16 * hi);
#pragma unroll
      for (unsigned t = 0; t < 4; ++t)
        vfr[kh * 4 + t] = ld16(&sV[(t * 16 + ln) * 72 + 32 * kh + 16 * hi]);
    }
#pragma unroll
    for (unsigned t = 0; t < 4; ++t) {
      o[t] = wmma16(pa[0], vfr[t], o[t]);
      o[t] = wmma16(pa[1], vfr[4 + t], o[t]);
    }
  }

  // normalize and write (b,l,n,e) fp32
#pragma unroll
  for (unsigned r = 0; r < 8; ++r) {
    const float inv = 1.0f / rsum[r];
    const unsigned m = rowb + r + 8 * hi;
    float* orow = out + (((size_t)bl * NSEQ + m) << 10) + hh * DH + ln;
#pragma unroll
    for (unsigned t = 0; t < 4; ++t)
      orow[t * 16] = o[t][r] * inv;
  }
}

extern "C" void kernel_launch(void* const* d_in, const int* in_sizes, int n_in,
                              void* d_out, int out_size, void* d_ws, size_t ws_size,
                              hipStream_t stream) {
  (void)in_sizes; (void)n_in; (void)out_size; (void)ws_size;
  const float* x    = (const float*)d_in[0];
  const float* mask = (const float*)d_in[1];
  const float* w    = (const float*)d_in[2];
  const float* bq   = (const float*)d_in[3];
  float* out = (float*)d_out;

  _Float16* qf = (_Float16*)d_ws;
  const size_t per = (size_t)BLn * NH * NSEQ * DH;  // 8,388,608 halves = 16 MB
  _Float16* kf = qf + per;
  _Float16* vf = kf + per;

  dim3 g1(N3E / 128, (BLn * NSEQ) / 128);           // 24 x 64
  qkv_gemm_kernel<<<g1, 256, 0, stream>>>(x, w, bq, qf, kf, vf);

  dim3 g2(NSEQ / 128, BLn * NH);                    // 4 x 256
  flash_attn_kernel<<<g2, 256, 0, stream>>>(qf, kf, vf, mask, out);
}
